// MultHeadsSelfAttentionOptQKV_80418967650597
// MI455X (gfx1250) — matrix operand order repaired and checked by reading, not compiled, not run on hardware
//
#include <hip/hip_runtime.h>
#include <hip/hip_bf16.h>

typedef __bf16 bf16_t;
typedef __attribute__((ext_vector_type(16))) __bf16 v16bf;
typedef __attribute__((ext_vector_type(8)))  __bf16 v8bf;
typedef __attribute__((ext_vector_type(4)))  __bf16 v4bf;
typedef __attribute__((ext_vector_type(8)))  float   v8f;

#define B_DIM 2
#define T_DIM 2048
#define K_DIM 1024
#define H_DIM 16
#define S_DIM 64

// LDS row stride (elements): 56 bf16 = 112B -> multiple of 16B (b128-aligned rows)
// and 28 dwords/row -> 16 consecutive rows hit 16 disjoint 4-bank groups.
#define LDSTR 56

union Frag16 { v16bf v; v8bf h[2]; };

__device__ __forceinline__ v8f wmma_bf16(const Frag16& a, const Frag16& b, v8f c) {
  // D = A(16x32 bf16) * B(32x16 bf16) + C(16x16 f32)
  return __builtin_amdgcn_wmma_f32_16x16x32_bf16(
      /*neg_a=*/false, a.v, /*neg_b=*/false, b.v,
      /*c_mod=*/(short)0, c, /*reuse_a=*/false, /*reuse_b=*/false);
}

__device__ __forceinline__ v4bf cvt4(float4 f) {
  v4bf r; r[0] = (bf16_t)f.x; r[1] = (bf16_t)f.y; r[2] = (bf16_t)f.z; r[3] = (bf16_t)f.w;
  return r;
}

// ---- DPP16 butterfly reductions within each 16-lane half (VALU-only, no DS) ----
// xor masks {1,2,7,15}: quad_perm(1,0,3,2)=0xB1, quad_perm(2,3,0,1)=0x4E,
// ROW_HALF_MIRROR=0x141 (l^7 in 8), ROW_MIRROR=0x140 (l^15 in 16).
template <int CTRL>
__device__ __forceinline__ float dppmov(float x) {
  return __int_as_float(
      __builtin_amdgcn_update_dpp(0, __float_as_int(x), CTRL, 0xF, 0xF, true));
}
__device__ __forceinline__ float rowmax16(float x) {
  x = fmaxf(x, dppmov<0xB1>(x));
  x = fmaxf(x, dppmov<0x4E>(x));
  x = fmaxf(x, dppmov<0x141>(x));
  x = fmaxf(x, dppmov<0x140>(x));
  return x;
}
__device__ __forceinline__ float rowsum16(float x) {
  x += dppmov<0xB1>(x);
  x += dppmov<0x4E>(x);
  x += dppmov<0x141>(x);
  x += dppmov<0x140>(x);
  return x;
}

// ---------------------------------------------------------------------------
// Projection GEMM: out = (X[M,1024] @ W[1024,1024]^T) * oscale, bf16 out to
//   VMODE==0: [b, h, t, s]   (Q with oscale=1/sqrt(k) folded in, K)
//   VMODE==1: [b, h, s, t]   (V, transposed so PV B-frags are contiguous)
// Block: 256 thr (8 waves), tile 128(M) x 64(N), K-step 32. Wave = 32x32.
// ---------------------------------------------------------------------------
template <int VMODE>
__global__ __launch_bounds__(256) void proj_kernel(const float* __restrict__ X,
                                                   const float* __restrict__ W,
                                                   bf16_t* __restrict__ out,
                                                   const float oscale) {
  __shared__ __align__(16) bf16_t lA[128][LDSTR];
  __shared__ __align__(16) bf16_t lB[64][LDSTR];

  const int tid = threadIdx.x;
  const int lane = tid & 31, wave = tid >> 5;
  const int wm = wave & 3, wn = wave >> 2;       // 4x2 wave grid
  const int blockM = blockIdx.x * 128, blockN = blockIdx.y * 64;
  const int lh = lane >> 4, col = lane & 15;     // lane half / lane column

  const int r0 = tid >> 3;                       // 0..31
  const int c4 = (tid & 7) * 4;                  // 0..28

  v8f acc[2][2];
#pragma unroll
  for (int i = 0; i < 2; ++i)
#pragma unroll
    for (int j = 0; j < 2; ++j)
      acc[i][j] = (v8f){0.f, 0.f, 0.f, 0.f, 0.f, 0.f, 0.f, 0.f};

  for (int kb = 0; kb < K_DIM; kb += 32) {
    // Stage A tile 128x32 f32 -> bf16 LDS
#pragma unroll
    for (int rr = 0; rr < 4; ++rr) {
      const int row = r0 + rr * 32;
      const float4 f = *(const float4*)(X + (size_t)(blockM + row) * K_DIM + kb + c4);
      *(v4bf*)&lA[row][c4] = cvt4(f);
    }
    // Stage B tile 64x32 f32 -> bf16 LDS (B row n = output feature n of W)
#pragma unroll
    for (int rr = 0; rr < 2; ++rr) {
      const int row = r0 + rr * 32;
      const float4 f = *(const float4*)(W + (size_t)(blockN + row) * K_DIM + kb + c4);
      *(v4bf*)&lB[row][c4] = cvt4(f);
    }
    if (kb + 32 < K_DIM) {  // global_prefetch_b8 for next K tile
      __builtin_prefetch(X + (size_t)(blockM + r0) * K_DIM + kb + 32 + c4, 0, 1);
      __builtin_prefetch(W + (size_t)(blockN + (r0 & 63)) * K_DIM + kb + 32 + c4, 0, 1);
    }
    __syncthreads();

    // A fragments (16x32): lane row = col; K elems {lh*8+0..7, 16+lh*8+0..7}
    Frag16 a[2], b[2];
#pragma unroll
    for (int i = 0; i < 2; ++i) {
      const bf16_t* p = &lA[wm * 32 + i * 16 + col][lh * 8];
      a[i].h[0] = *(const v8bf*)p;
      a[i].h[1] = *(const v8bf*)(p + 16);
    }
    // B fragments (32x16): lane col n = col; K elems {lh*8+0..7, 16+lh*8+0..7}
#pragma unroll
    for (int j = 0; j < 2; ++j) {
      const bf16_t* p = &lB[wn * 32 + j * 16 + col][lh * 8];
      b[j].h[0] = *(const v8bf*)p;
      b[j].h[1] = *(const v8bf*)(p + 16);
    }
#pragma unroll
    for (int i = 0; i < 2; ++i)
#pragma unroll
      for (int j = 0; j < 2; ++j)
        acc[i][j] = wmma_bf16(a[i], b[j], acc[i][j]);
    __syncthreads();
  }

  // C layout: VGPR r, lanes 0-15 -> M=r, lanes 16-31 -> M=r+8; N = col
#pragma unroll
  for (int i = 0; i < 2; ++i)
#pragma unroll
    for (int j = 0; j < 2; ++j)
#pragma unroll
      for (int r = 0; r < 8; ++r) {
        const int m = blockM + wm * 32 + i * 16 + r + 8 * lh;
        const int n = blockN + wn * 32 + j * 16 + col;
        const int bb = m >> 11, t = m & (T_DIM - 1);
        const int h = n >> 6, s = n & (S_DIM - 1);
        const size_t idx =
            (VMODE == 0) ? ((size_t)((bb * H_DIM + h) * T_DIM + t) * S_DIM + s)
                         : ((size_t)((bb * H_DIM + h) * S_DIM + s) * T_DIM + t);
        out[idx] = (bf16_t)(acc[i][j][r] * oscale);
      }
}

// ---------------------------------------------------------------------------
// Flash attention: one wave owns (b, h, 16-query tile). Online softmax over
// 32-key chunks; scores and PV both via bf16 WMMA. q (pre-scaled by 1/sqrt(k))
// and k in [b,h,t,s]; v transposed [b,h,s,t]; attn out bf16 in [b,t,h*s].
// ---------------------------------------------------------------------------
__global__ __launch_bounds__(128) void attn_kernel(const bf16_t* __restrict__ q,
                                                   const bf16_t* __restrict__ k,
                                                   const bf16_t* __restrict__ vt,
                                                   bf16_t* __restrict__ attn_out) {
  __shared__ __align__(16) bf16_t Pt[4][16][LDSTR];  // per-wave P tile (16x32)

  const int tid = threadIdx.x, lane = tid & 31, w = tid >> 5;
  const int gw = blockIdx.x * 4 + w;     // 0..4095
  const int bh = gw >> 7;                // 0..31  (b*16 + h)
  const int qtile = gw & 127;
  const int qbase = qtile * 16;
  const int lh = lane >> 4, col = lane & 15;

  const bf16_t* qp = q + (size_t)bh * T_DIM * S_DIM;
  const bf16_t* kp = k + (size_t)bh * T_DIM * S_DIM;
  const bf16_t* vp = vt + (size_t)bh * S_DIM * T_DIM;

  // Q fragments: rows qbase..qbase+15, K = s in two 32-wide chunks
  Frag16 qf[2];
  {
    const bf16_t* qrow = qp + (size_t)(qbase + col) * S_DIM;
#pragma unroll
    for (int ks = 0; ks < 2; ++ks) {
      const bf16_t* p = qrow + ks * 32 + lh * 8;
      qf[ks].h[0] = *(const v8bf*)p;
      qf[ks].h[1] = *(const v8bf*)(p + 16);
    }
  }

  float mrow[8], lrow[8];
  v8f o[4];
#pragma unroll
  for (int r = 0; r < 8; ++r) { mrow[r] = -1e30f; lrow[r] = 0.f; }
#pragma unroll
  for (int j = 0; j < 4; ++j) o[j] = (v8f){0.f, 0.f, 0.f, 0.f, 0.f, 0.f, 0.f, 0.f};

  for (int tc = 0; tc < T_DIM; tc += 32) {
    // ---- scores S[16 x 32] = q @ k^T for 2 column tiles of 16 ----
    v8f s[2];
#pragma unroll
    for (int tt = 0; tt < 2; ++tt) {
      const bf16_t* krow = kp + (size_t)(tc + tt * 16 + col) * S_DIM;
      Frag16 kf0, kf1;  // B frags: column n = key T, K = s
      kf0.h[0] = *(const v8bf*)(krow + lh * 8);
      kf0.h[1] = *(const v8bf*)(krow + lh * 8 + 16);
      kf1.h[0] = *(const v8bf*)(krow + 32 + lh * 8);
      kf1.h[1] = *(const v8bf*)(krow + 32 + lh * 8 + 16);
      v8f z = (v8f){0.f, 0.f, 0.f, 0.f, 0.f, 0.f, 0.f, 0.f};
      z = wmma_bf16(qf[0], kf0, z);
      z = wmma_bf16(qf[1], kf1, z);
      s[tt] = z;
    }
    // ---- issue V B-frag loads early so they overlap the softmax VALU work ----
    Frag16 vf[4];
#pragma unroll
    for (int j = 0; j < 4; ++j) {
      const bf16_t* vrow = vp + (size_t)(j * 16 + col) * T_DIM + tc + lh * 8;
      vf[j].h[0] = *(const v8bf*)vrow;
      vf[j].h[1] = *(const v8bf*)(vrow + 16);
    }
    // ---- online softmax; each C-row lives in one 16-lane half (DPP reduce) ----
#pragma unroll
    for (int r = 0; r < 8; ++r) {
      const float x0 = s[0][r], x1 = s[1][r];   // scale pre-folded into q
      const float cm = rowmax16(fmaxf(x0, x1));
      const float nm = fmaxf(mrow[r], cm);
      const float p0 = __expf(x0 - nm), p1 = __expf(x1 - nm);
      const float rs = rowsum16(p0 + p1);
      const float alpha = __expf(mrow[r] - nm);
      lrow[r] = lrow[r] * alpha + rs;
      mrow[r] = nm;
      const int prow = r + 8 * lh;
      Pt[w][prow][col] = (bf16_t)p0;
      Pt[w][prow][16 + col] = (bf16_t)p1;
#pragma unroll
      for (int j = 0; j < 4; ++j) o[j][r] *= alpha;
    }
    // ---- P (16x32) A-fragment from LDS; accumulate O += P @ V ----
    Frag16 pf;
    {
      const bf16_t* p = &Pt[w][col][lh * 8];
      pf.h[0] = *(const v8bf*)p;
      pf.h[1] = *(const v8bf*)(p + 16);
    }
#pragma unroll
    for (int j = 0; j < 4; ++j) o[j] = wmma_bf16(pf, vf[j], o[j]);
  }

  // epilogue: normalize and store attn in [b, t, h*64 + s] bf16
  const int bb = bh >> 4, h = bh & 15;
  float inv[8];
#pragma unroll
  for (int r = 0; r < 8; ++r) inv[r] = 1.0f / lrow[r];
#pragma unroll
  for (int j = 0; j < 4; ++j)
#pragma unroll
    for (int r = 0; r < 8; ++r) {
      const int t = qbase + r + 8 * lh;
      const size_t idx = ((size_t)(bb * T_DIM + t)) * K_DIM + h * S_DIM + j * 16 + col;
      attn_out[idx] = (bf16_t)(o[j][r] * inv[r]);
    }
}

// ---------------------------------------------------------------------------
// Output projection: out[M,1024] = attn_bf16 @ Wu^T + bu  (f32 out).
// A frags loaded straight from global bf16; Wu staged via LDS (f32->bf16).
// ---------------------------------------------------------------------------
__global__ __launch_bounds__(256) void outproj_kernel(const bf16_t* __restrict__ Xb,
                                                      const float* __restrict__ W,
                                                      const float* __restrict__ bias,
                                                      float* __restrict__ out) {
  __shared__ __align__(16) bf16_t lB[64][LDSTR];

  const int tid = threadIdx.x;
  const int lane = tid & 31, wave = tid >> 5;
  const int wm = wave & 3, wn = wave >> 2;
  const int blockM = blockIdx.x * 128, blockN = blockIdx.y * 64;
  const int lh = lane >> 4, col = lane & 15;

  const int r0 = tid >> 3;
  const int c4 = (tid & 7) * 4;

  v8f acc[2][2];
#pragma unroll
  for (int i = 0; i < 2; ++i)
#pragma unroll
    for (int j = 0; j < 2; ++j)
      acc[i][j] = (v8f){0.f, 0.f, 0.f, 0.f, 0.f, 0.f, 0.f, 0.f};

  for (int kb = 0; kb < K_DIM; kb += 32) {
#pragma unroll
    for (int rr = 0; rr < 2; ++rr) {
      const int row = r0 + rr * 32;
      const float4 f = *(const float4*)(W + (size_t)(blockN + row) * K_DIM + kb + c4);
      *(v4bf*)&lB[row][c4] = cvt4(f);
    }
    if (kb + 32 < K_DIM)
      __builtin_prefetch(W + (size_t)(blockN + (r0 & 63)) * K_DIM + kb + 32 + c4, 0, 1);
    __syncthreads();

    Frag16 a[2], b[2];
#pragma unroll
    for (int i = 0; i < 2; ++i) {  // A direct from global bf16 (rows contiguous)
      const int m = blockM + wm * 32 + i * 16 + col;
      const bf16_t* p = Xb + (size_t)m * K_DIM + kb + lh * 8;
      a[i].h[0] = *(const v8bf*)p;
      a[i].h[1] = *(const v8bf*)(p + 16);
    }
#pragma unroll
    for (int j = 0; j < 2; ++j) {
      const bf16_t* p = &lB[wn * 32 + j * 16 + col][lh * 8];
      b[j].h[0] = *(const v8bf*)p;
      b[j].h[1] = *(const v8bf*)(p + 16);
    }
#pragma unroll
    for (int i = 0; i < 2; ++i)
#pragma unroll
      for (int j = 0; j < 2; ++j)
        acc[i][j] = wmma_bf16(a[i], b[j], acc[i][j]);
    __syncthreads();
  }

#pragma unroll
  for (int i = 0; i < 2; ++i)
#pragma unroll
    for (int j = 0; j < 2; ++j) {
      const int n = blockN + wn * 32 + j * 16 + col;
      const float bv = bias[n];
#pragma unroll
      for (int r = 0; r < 8; ++r) {
        const int m = blockM + wm * 32 + i * 16 + r + 8 * lh;
        out[(size_t)m * K_DIM + n] = acc[i][j][r] + bv;
      }
    }
}

extern "C" void kernel_launch(void* const* d_in, const int* in_sizes, int n_in,
                              void* d_out, int out_size, void* d_ws, size_t ws_size,
                              hipStream_t stream) {
  const float* Q  = (const float*)d_in[0];
  const float* K  = (const float*)d_in[1];
  const float* V  = (const float*)d_in[2];
  const float* Wq = (const float*)d_in[3];
  const float* Wk = (const float*)d_in[4];
  const float* Wv = (const float*)d_in[5];
  const float* Wu = (const float*)d_in[6];
  const float* bu = (const float*)d_in[7];
  float* out = (float*)d_out;

  bf16_t* ws = (bf16_t*)d_ws;
  const size_t per = (size_t)B_DIM * H_DIM * T_DIM * S_DIM;  // 4Mi elems (8MB)
  bf16_t* qws = ws;
  bf16_t* kws = ws + per;
  bf16_t* vws = ws + 2 * per;
  bf16_t* aws = ws + 3 * per;

  const dim3 gproj(32, 16), b256(256);
  // fold 1/sqrt(1024) into the Q projection output
  proj_kernel<0><<<gproj, b256, 0, stream>>>(Q, Wq, qws, 0.03125f);
  proj_kernel<0><<<gproj, b256, 0, stream>>>(K, Wk, kws, 1.0f);
  proj_kernel<1><<<gproj, b256, 0, stream>>>(V, Wv, vws, 1.0f);
  attn_kernel<<<dim3(1024), dim3(128), 0, stream>>>(qws, kws, vws, aws);
  outproj_kernel<<<gproj, b256, 0, stream>>>(aws, Wu, bu, out);
}
